// InvResNet_80401787781415
// MI455X (gfx1250) — matrix operand order repaired and checked by reading, not compiled, not run on hardware
//
#include <hip/hip_runtime.h>
#include <hip/hip_bf16.h>

// Problem constants (match reference)
#define B_   16
#define N_   20000
#define C_   64
#define DIM_ 128
#define KC_  3
#define E_   640000
#define NB_  3

typedef __attribute__((ext_vector_type(16))) __bf16 v16bf;
typedef __attribute__((ext_vector_type(8)))  float  v8f;

// ---- WMMA LDS layout helpers (CDNA5 wave32, 16-bit operands) ----
// A-matrix 16x32 (MxK): lanes 0-15 hold row M=lane with K in {0..7,16..23},
// lanes 16-31 hold row M=lane-16 with K in {8..15,24..31}.  We store each
// lane's 16 bf16 contiguously so the wave load is one v16bf per lane.
__device__ __forceinline__ int aIdx(int m, int k) {  // k in [0,32)
    int lane = m + 16 * ((k >> 3) & 1);
    int elem = (k & 7) + ((k >> 4) << 3);
    return lane * 16 + elem;
}
// B-matrix 32x16 (KxN): lanes 0-15 hold col N=lane, K in {0..7,16..23}; lanes 16-31 hold
// col N=lane-16, K in {8..15,24..31}.  16 contiguous bf16 per lane.
__device__ __forceinline__ int bIdx(int k, int n) {  // k in [0,32), n in [0,16)
    int lane = n + 16 * ((k >> 3) & 1);
    int elem = (k & 7) + ((k >> 4) << 3);
    return lane * 16 + elem;
}

__device__ __forceinline__ float swish_f(float x, float sb) {
    // x * sigmoid(x*softplus(beta)) / 1.1 ; sb = softplus(beta)
    return x * (1.0f / (1.0f + __expf(-x * sb))) * (1.0f / 1.1f);
}

// ---------------- degree / edge-norm ----------------
__global__ void deg_kernel(const int* __restrict__ src,
                           const float* __restrict__ ew,
                           float* __restrict__ deg) {
    int e = blockIdx.x * 256 + threadIdx.x;
    if (e < E_) atomicAdd(&deg[src[e]], ew[e]);
}

__global__ void wnorm_kernel(const int* __restrict__ src,
                             const int* __restrict__ dst,
                             const float* __restrict__ ew,
                             const float* __restrict__ deg,
                             float* __restrict__ wn) {
    int e = blockIdx.x * 256 + threadIdx.x;
    if (e >= E_) return;
    float ds = deg[src[e]], dd = deg[dst[e]];
    float is = ds > 0.f ? __frsqrt_rn(fmaxf(ds, 1e-12f)) : 0.f;
    float id = dd > 0.f ? __frsqrt_rn(fmaxf(dd, 1e-12f)) : 0.f;
    wn[e] = -is * ew[e] * id;
}

// ---------------- weight repack f32 -> bf16 in WMMA B-layout ----------------
#define NCHEB (NB_ * 3 * 64 * 128)
#define NW1   (NB_ * 128 * 128)
#define NW2   (NB_ * 128 * 64)

__global__ void prep_weights(const float* __restrict__ cheb_w,
                             const float* __restrict__ w1,
                             const float* __restrict__ w2,
                             __bf16* __restrict__ chebWb,  // [NB*3][2 kc][8 cs][512]
                             __bf16* __restrict__ w1b,     // [NB][4 kc][8 cs][512]
                             __bf16* __restrict__ w2b) {   // [NB][4 kc][4 cs][512]
    int i = blockIdx.x * 256 + threadIdx.x;
    if (i < NCHEB) {
        int n = i & 127, k = (i >> 7) & 63, bt = i >> 13;           // (b*3+t, k, n)
        chebWb[bt * (2 * 8 * 512) + ((k >> 5) * 8 + (n >> 4)) * 512 + bIdx(k & 31, n & 15)]
            = (__bf16)cheb_w[i];
    } else if (i < NCHEB + NW1) {
        int j = i - NCHEB;
        int n = j & 127, k = (j >> 7) & 127, b = j >> 14;
        w1b[b * (4 * 8 * 512) + ((k >> 5) * 8 + (n >> 4)) * 512 + bIdx(k & 31, n & 15)]
            = (__bf16)w1[j];
    } else if (i < NCHEB + NW1 + NW2) {
        int j = i - NCHEB - NW1;
        int n = j & 63, k = (j >> 6) & 127, b = j >> 13;
        w2b[b * (4 * 4 * 512) + ((k >> 5) * 4 + (n >> 4)) * 512 + bIdx(k & 31, n & 15)]
            = (__bf16)w2[j];
    }
}

// ---------------- SpMM scatter: accum[b,dst,:] += scale*w[e]*h[b,src,:] ----------------
__global__ __launch_bounds__(256)
void spmm_scatter(const int* __restrict__ srcIdx, const int* __restrict__ dstIdx,
                  const float* __restrict__ wn, const float* __restrict__ hin,
                  float* __restrict__ accum, float scale) {
    long wave = ((long)blockIdx.x * 256 + threadIdx.x) >> 5;
    int lane = threadIdx.x & 31;
    if (wave >= (long)E_ * B_) return;
    int e = (int)(wave / B_);
    int b = (int)(wave % B_);
    float w = wn[e] * scale;
    int s = srcIdx[e], d = dstIdx[e];
    const float2* hp = (const float2*)(hin + ((long)b * N_ + (long)s) * C_);
    float2 v = hp[lane];                               // 32 lanes x float2 = 64 ch
    float* ap = accum + ((long)b * N_ + (long)d) * C_ + lane * 2;
    atomicAdd(ap,     w * v.x);
    atomicAdd(ap + 1, w * v.y);
}

// ---------------- fused Cheb-combine + 2-layer MLP + residual ----------------
// One 16-row tile per workgroup; 8 waves own 16-column output slices.
__global__ __launch_bounds__(256)
void cheb_mlp_kernel(float* __restrict__ h,            // [B*N, 64] in/out (running state)
                     const float* __restrict__ t1,     // [B*N, 64]
                     const float* __restrict__ t2raw,  // [B*N, 64] = 2*spmm(t1)
                     const __bf16* __restrict__ chebWb,// this block: [3][2][8][512]
                     const __bf16* __restrict__ w1b,   // this block: [4][8][512]
                     const __bf16* __restrict__ w2b,   // this block: [4][4][512]
                     const float* __restrict__ chebBias, // [128]
                     const float* __restrict__ b1v,      // [128]
                     const float* __restrict__ b2v,      // [64]
                     const float* __restrict__ betaPtr) {
    __shared__ __bf16 ldsA[3][2][512];  // T0/T1/T2 tiles, A-layout, 2 K-chunks of 32
    __shared__ __bf16 ldsG[4][512];     // 16x128 activation, A-layout, 4 K-chunks

    const int  tid  = threadIdx.x;
    const int  lane = tid & 31;
    const int  wv   = tid >> 5;
    const long rowBase = (long)blockIdx.x * 16;
    const float sb = log1pf(__expf(*betaPtr));   // softplus(beta)

    // Stage T0, T1, T2(=t2raw - T0) tiles into LDS as bf16 in A-layout.
    for (int i = tid; i < 3 * 16 * 64; i += 256) {
        int t   = i >> 10;
        int rem = i & 1023;
        int m   = rem >> 6;
        int k   = rem & 63;
        long g  = (rowBase + m) * C_ + k;
        float v;
        if (t == 0)      v = h[g];
        else if (t == 1) v = t1[g];
        else             v = t2raw[g] - h[g];
        ldsA[t][k >> 5][aIdx(m, k & 31)] = (__bf16)v;
    }
    __syncthreads();

    // GEMM1: out[16x128] = sum_t Tt[16x64] @ Wt[64x128]; wave wv -> cols 16wv..16wv+15
    v8f acc = {};
    #pragma unroll
    for (int t = 0; t < 3; ++t)
        #pragma unroll
        for (int c = 0; c < 2; ++c) {
            v16bf A  = *(const v16bf*)&ldsA[t][c][lane * 16];
            v16bf Bm = *(const v16bf*)&chebWb[((t * 2 + c) * 8 + wv) * 512 + lane * 16];
            acc = __builtin_amdgcn_wmma_f32_16x16x32_bf16(false, A, false, Bm,
                                                          (short)0, acc, false, false);
        }
    {   // bias + swish; write g into LDS A-layout (col => K of next GEMM)
        int nGlob = (wv << 4) + (lane & 15);
        float bias = chebBias[nGlob];
        int cg = nGlob >> 5, kk = nGlob & 31, mh = (lane >> 4) << 3;
        #pragma unroll
        for (int v = 0; v < 8; ++v) {
            float gv = swish_f(acc[v] + bias, sb);
            ldsG[cg][aIdx(v + mh, kk)] = (__bf16)gv;
        }
    }
    __syncthreads();

    // GEMM2: g[16x128] @ w1[128x128]
    v8f acc2 = {};
    #pragma unroll
    for (int c = 0; c < 4; ++c) {
        v16bf A  = *(const v16bf*)&ldsG[c][lane * 16];
        v16bf Bm = *(const v16bf*)&w1b[(c * 8 + wv) * 512 + lane * 16];
        acc2 = __builtin_amdgcn_wmma_f32_16x16x32_bf16(false, A, false, Bm,
                                                       (short)0, acc2, false, false);
    }
    __syncthreads();   // all waves done reading ldsG before overwrite
    {
        int nGlob = (wv << 4) + (lane & 15);
        float bias = b1v[nGlob];
        int cg = nGlob >> 5, kk = nGlob & 31, mh = (lane >> 4) << 3;
        #pragma unroll
        for (int v = 0; v < 8; ++v) {
            float gv = swish_f(acc2[v] + bias, sb);
            ldsG[cg][aIdx(v + mh, kk)] = (__bf16)gv;
        }
    }
    __syncthreads();

    // GEMM3: g2[16x128] @ w2[128x64]; waves 0..3 cover the 64 output cols
    if (wv < 4) {
        v8f acc3 = {};
        #pragma unroll
        for (int c = 0; c < 4; ++c) {
            v16bf A  = *(const v16bf*)&ldsG[c][lane * 16];
            v16bf Bm = *(const v16bf*)&w2b[(c * 4 + wv) * 512 + lane * 16];
            acc3 = __builtin_amdgcn_wmma_f32_16x16x32_bf16(false, A, false, Bm,
                                                           (short)0, acc3, false, false);
        }
        int n = (wv << 4) + (lane & 15);
        float bias = b2v[n];
        int mh = (lane >> 4) << 3;
        #pragma unroll
        for (int v = 0; v < 8; ++v) {
            long g = (rowBase + v + mh) * C_ + n;
            h[g] = h[g] + acc3[v] + bias;   // invertible residual, in place
        }
    }
}

// ---------------- host-side orchestration ----------------
extern "C" void kernel_launch(void* const* d_in, const int* in_sizes, int n_in,
                              void* d_out, int out_size, void* d_ws, size_t ws_size,
                              hipStream_t stream) {
    const float* x      = (const float*)d_in[0];
    const int*   ei     = (const int*)d_in[1];     // [2, E]
    const float* ew     = (const float*)d_in[2];
    const float* cheb_w = (const float*)d_in[3];
    const float* cheb_b = (const float*)d_in[4];
    const float* beta   = (const float*)d_in[5];
    const float* w1     = (const float*)d_in[6];
    const float* b1     = (const float*)d_in[7];
    const float* w2     = (const float*)d_in[8];
    const float* b2     = (const float*)d_in[9];
    float* h = (float*)d_out;                       // running state lives in d_out

    const size_t HBYTES = (size_t)B_ * N_ * C_ * sizeof(float);
    char* ws = (char*)d_ws;
    size_t off = 0;
    auto carve = [&](size_t bytes) {
        void* p = ws + off;
        off = (off + bytes + 255) & ~(size_t)255;
        return p;
    };
    float*  t1     = (float*)carve(HBYTES);
    float*  t2     = (float*)carve(HBYTES);
    float*  wn     = (float*)carve((size_t)E_ * sizeof(float));
    float*  deg    = (float*)carve((size_t)N_ * sizeof(float));
    __bf16* chebWb = (__bf16*)carve((size_t)NCHEB * sizeof(__bf16));
    __bf16* w1b    = (__bf16*)carve((size_t)NW1 * sizeof(__bf16));
    __bf16* w2b    = (__bf16*)carve((size_t)NW2 * sizeof(__bf16));
    (void)ws_size; (void)in_sizes; (void)n_in; (void)out_size;

    const int* srcIdx = ei;
    const int* dstIdx = ei + E_;

    // h = x
    hipMemcpyAsync(h, x, HBYTES, hipMemcpyDeviceToDevice, stream);

    // edge normalization
    hipMemsetAsync(deg, 0, (size_t)N_ * sizeof(float), stream);
    deg_kernel<<<(E_ + 255) / 256, 256, 0, stream>>>(srcIdx, ew, deg);
    wnorm_kernel<<<(E_ + 255) / 256, 256, 0, stream>>>(srcIdx, dstIdx, ew, deg, wn);

    // weights -> bf16, WMMA B-layout
    prep_weights<<<(NCHEB + NW1 + NW2 + 255) / 256, 256, 0, stream>>>(
        cheb_w, w1, w2, chebWb, w1b, w2b);

    const long waves = (long)E_ * B_;
    const int spmmBlocks = (int)((waves * 32 + 255) / 256);
    const int tiles = (B_ * N_) / 16;

    for (int b = 0; b < NB_; ++b) {
        hipMemsetAsync(t1, 0, HBYTES, stream);
        spmm_scatter<<<spmmBlocks, 256, 0, stream>>>(srcIdx, dstIdx, wn, h, t1, 1.0f);
        hipMemsetAsync(t2, 0, HBYTES, stream);
        spmm_scatter<<<spmmBlocks, 256, 0, stream>>>(srcIdx, dstIdx, wn, t1, t2, 2.0f);
        cheb_mlp_kernel<<<tiles, 256, 0, stream>>>(
            h, t1, t2,
            chebWb + (size_t)b * 3 * (2 * 8 * 512),
            w1b    + (size_t)b * (4 * 8 * 512),
            w2b    + (size_t)b * (4 * 4 * 512),
            cheb_b + (size_t)b * 128,
            b1     + (size_t)b * 128,
            b2     + (size_t)b * 64,
            beta + b);
    }
}
